// SpikingUnifiedCoreFlow_46686294507989
// MI455X (gfx1250) — matrix operand order repaired and checked by reading, not compiled, not run on hardware
//
#include <hip/hip_runtime.h>
#include <math.h>

// ---------------------------------------------------------------------------
// SNN rate-mode simulation for MI455X (gfx1250), wave32 + WMMA f16.
//   x:       (128, 1024) f32 in [0,1]
//   weights: (4, 1024, 1024) f32  (neurons, axons)
//   thr:     (4,) f32
//   out:     (128, 1024) f32
// Strategy: batch-tiled persistent workgroups (16 rows each, M=16 WMMA tile),
// membranes in VGPRs, spikes ping-ponged through LDS, f16 weights resident in
// L2, v_wmma_f32_16x16x32_f16 for all matmuls.
// ---------------------------------------------------------------------------

typedef _Float16 v16h __attribute__((ext_vector_type(16)));
typedef _Float16 h8   __attribute__((ext_vector_type(8)));
typedef float    v8f  __attribute__((ext_vector_type(8)));

#define T_CYC  32
#define NLAYER 4
#define HDIM   1024
#define BTILE  16
#define SROW   1032      // padded LDS row stride (halves) to spread banks
#define NT     8         // 16-wide N tiles per wave (8 waves * 8 * 16 = 1024)

// ---- prep: convert f32 weights -> f16 copy in workspace (L2-resident) -----
__global__ void snn_wconv(const float* __restrict__ w,
                          _Float16* __restrict__ wh) {
    int i = blockIdx.x * 256 + threadIdx.x;
    wh[i] = (_Float16)w[i];
}

// ---- main: whole 32-cycle, 4-layer chain per batch tile --------------------
__launch_bounds__(256, 1)
__global__ void snn_main(const float* __restrict__ x,
                         const _Float16* __restrict__ wh,
                         const float* __restrict__ thr,
                         float* __restrict__ out) {
    __shared__ unsigned  s_mask[BTILE * HDIM];        // 64 KB spike bitmasks
    __shared__ _Float16  s_sig[2][BTILE * SROW];      // 2 x 33 KB spike f16

    const int tid    = threadIdx.x;
    const int lane   = tid & 31;
    const int wave   = tid >> 5;
    const int laneLo = lane & 15;
    const int laneHi = lane >> 4;
    const int wg     = blockIdx.x;                    // batch tile 0..7

    // ---- one-time uniform-spike encoding: 32-bit fire mask per element ----
    for (int idx = tid; idx < BTILE * HDIM; idx += 256) {
        float xv = x[wg * BTILE * HDIM + idx];
        int   N  = (int)rintf(xv * 32.0f);
        unsigned mk = 0u;
        if (N >= T_CYC) {
            mk = 0xFFFFFFFFu;                         // saturated: fire always
        } else if (N > 0) {
            float spacing = 32.0f / (float)N;
            for (int c = 0; c < T_CYC; ++c) {
                float cf = (float)c;
                bool fire = (floorf(cf / spacing) < (float)N) &&
                            (floorf(fmodf(cf, spacing)) == 0.0f);
                mk |= fire ? (1u << c) : 0u;
            }
        }
        s_mask[idx] = mk;
    }

    // ---- per-wave accumulator state (registers) ---------------------------
    v8f mem[NLAYER][NT];                              // membranes: 256 VGPRs
    v8f accv[NT];                                     // spike-count accum
#pragma unroll
    for (int l = 0; l < NLAYER; ++l)
#pragma unroll
        for (int t = 0; t < NT; ++t)
#pragma unroll
            for (int r = 0; r < 8; ++r) mem[l][t][r] = 0.0f;
#pragma unroll
    for (int t = 0; t < NT; ++t)
#pragma unroll
        for (int r = 0; r < 8; ++r) accv[t][r] = 0.0f;

    // A fragment base: 16-bit A 16x32 layout -> per lane two 16B runs
    const _Float16* abase0 = &s_sig[0][laneLo * SROW + laneHi * 8];
    const _Float16* abase1 = &s_sig[1][laneLo * SROW + laneHi * 8];
    // B fragment base: 16-bit B 32x16 layout -> per lane two 16B runs
    const long bn = (long)((wave * NT) * 16 + laneLo) * HDIM + laneHi * 8;

    for (int c = 0; c < T_CYC; ++c) {
        __syncthreads();
        // layer-0 input spikes for this cycle from the bitmask
        for (int idx = tid; idx < BTILE * HDIM; idx += 256) {
            int m = idx >> 10, k = idx & (HDIM - 1);
            s_sig[0][m * SROW + k] =
                (_Float16)(float)((s_mask[idx] >> c) & 1u);
        }
        __syncthreads();

#pragma unroll
        for (int l = 0; l < NLAYER; ++l) {
            const _Float16* wl = wh + (long)l * (HDIM * HDIM);
            const int cur = l & 1;                    // read buffer
            const _Float16* ab = (cur == 0) ? abase0 : abase1;
            const _Float16* bb = wl + bn;

            // warm next layer's weights in L2 (global_prefetch_b8)
            __builtin_prefetch(
                (const void*)(wh + (long)((l + 1) & 3) * (HDIM * HDIM) +
                              (long)tid * 2048), 0, 1);

            // ---- sig @ W^T via v_wmma_f32_16x16x32_f16 --------------------
            for (int kt = 0; kt < 32; ++kt) {
                union { v16h v; h8 h[2]; } a;
                a.h[0] = *(const h8*)(ab + kt * 32);
                a.h[1] = *(const h8*)(ab + kt * 32 + 16);
#pragma unroll
                for (int nt = 0; nt < NT; ++nt) {
                    union { v16h v; h8 h[2]; } b;
                    b.h[0] = *(const h8*)(bb + (long)nt * 16 * HDIM + kt * 32);
                    b.h[1] = *(const h8*)(bb + (long)nt * 16 * HDIM + kt * 32 + 16);
                    mem[l][nt] = __builtin_amdgcn_wmma_f32_16x16x32_f16(
                        false, a.v, false, b.v, (short)0, mem[l][nt],
                        false, false);
                }
            }

            // ---- LIF: spike + subtract reset; forward spikes --------------
            const float tl = thr[l];
#pragma unroll
            for (int nt = 0; nt < NT; ++nt) {
                const int ncol = (wave * NT + nt) * 16 + laneLo;
#pragma unroll
                for (int r = 0; r < 8; ++r) {
                    float mv = mem[l][nt][r];
                    float s  = (tl < mv) ? 1.0f : 0.0f;
                    mem[l][nt][r] = mv - s * tl;
                    if (l < NLAYER - 1) {
                        s_sig[cur ^ 1][(r + laneHi * 8) * SROW + ncol] =
                            (_Float16)s;
                    } else {
                        accv[nt][r] += s;             // last core spike count
                    }
                }
            }
            __syncthreads();
        }
    }

    // ---- write rates: mean over T cycles ----------------------------------
    const float inv = 1.0f / 32.0f;
#pragma unroll
    for (int nt = 0; nt < NT; ++nt) {
        const int ncol = (wave * NT + nt) * 16 + laneLo;
#pragma unroll
        for (int r = 0; r < 8; ++r) {
            out[(long)(wg * BTILE + r + laneHi * 8) * HDIM + ncol] =
                accv[nt][r] * inv;
        }
    }
}

extern "C" void kernel_launch(void* const* d_in, const int* in_sizes, int n_in,
                              void* d_out, int out_size, void* d_ws,
                              size_t ws_size, hipStream_t stream) {
    const float* x   = (const float*)d_in[0];   // (128,1024)
    const float* w   = (const float*)d_in[1];   // (4,1024,1024)
    const float* thr = (const float*)d_in[2];   // (4,)
    float*       out = (float*)d_out;           // (128,1024)
    _Float16*    wh  = (_Float16*)d_ws;         // 8 MB f16 weight copy

    const int nw = NLAYER * HDIM * HDIM;        // 4M elements
    snn_wconv<<<nw / 256, 256, 0, stream>>>(w, wh);
    snn_main<<<128 / BTILE, 256, 0, stream>>>(x, wh, thr, out);
}
